// HAMPLayer_43585328120460
// MI455X (gfx1250) — compile-verified, matrix operand order checked by reading, not run on hardware
//
#include <hip/hip_runtime.h>
#include <hip/hip_bf16.h>

// ---------------------------------------------------------------------------
// HAMP heterogeneous graph attention layer for gfx1250 (MI455X).
// D=256, H=8, DK=32. bf16 WMMA (v_wmma_f32_16x16x32_bf16) for all dense GEMMs,
// combined per-head transforms folded into the projection weights, atomics for
// the segmented softmax / aggregation, fused residual + LayerNorm epilogue.
// Weight staging into LDS uses CDNA5 async global->LDS copies (ASYNCcnt).
// ---------------------------------------------------------------------------

typedef unsigned int u32;
typedef unsigned long long u64;
typedef __attribute__((ext_vector_type(16))) __bf16 bf16x16;
typedef __attribute__((ext_vector_type(8)))  float  f32x8;

union FragBF { uint4 q[2]; bf16x16 v; };

__device__ __forceinline__ u32 pack_bf2(float a, float b) {
  u32 ua = __float_as_uint(a); ua += 0x7fffu + ((ua >> 16) & 1u);
  u32 ub = __float_as_uint(b); ub += 0x7fffu + ((ub >> 16) & 1u);
  return (ua >> 16) | (ub & 0xffff0000u);
}
// order-preserving float<->uint encoding for atomicMax-based segment max
__device__ __forceinline__ u32 enc_f32(float f) {
  u32 u = __float_as_uint(f);
  return (u & 0x80000000u) ? ~u : (u | 0x80000000u);
}
__device__ __forceinline__ float dec_f32(u32 u) {
  u32 b = (u & 0x80000000u) ? (u ^ 0x80000000u) : ~u;
  return __uint_as_float(b);
}

// ---------------------------------------------------------------------------
// Kernel 1: build 10 combined 256x256 weights (f32 scratch) + combined biases.
//  p0: Wq[0]            p1: Wq[1]
//  p2: Wk[0]*att0*c0/sdk  p3: Wv[0]*val0
//  p4: Wk[1]*att1*c1/sdk  p5: Wv[1]*val1
//  p6: Wk[1]*att2*c2/sdk  p7: Wv[1]*val2
//  p8: Wfc[0]           p9: Wfc[1]
// ---------------------------------------------------------------------------
__global__ __launch_bounds__(256)
void build_weights_kernel(const float* __restrict__ Wq, const float* __restrict__ bq,
                          const float* __restrict__ Wk, const float* __restrict__ bk,
                          const float* __restrict__ Wv, const float* __restrict__ bv,
                          const float* __restrict__ Wfc, const float* __restrict__ bfc,
                          const float* __restrict__ att, const float* __restrict__ val,
                          const float* __restrict__ canon,
                          float* __restrict__ Wtmp, float* __restrict__ bcomb) {
  const int p = blockIdx.x;
  const int n = threadIdx.x;           // output column 0..255
  const int h  = n >> 5;
  const int np = n & 31;
  const float inv_sdk = 0.17677669529663687f;  // 1/sqrt(32)

  const float* Ws = nullptr; const float* bs = nullptr;
  const float* T  = nullptr; float cs = 1.0f;
  switch (p) {
    case 0: Ws = Wq;          bs = bq;        break;
    case 1: Ws = Wq + 65536;  bs = bq + 256;  break;
    case 2: Ws = Wk;          bs = bk;        T = att;          cs = canon[h]      * inv_sdk; break;
    case 3: Ws = Wv;          bs = bv;        T = val;          break;
    case 4: Ws = Wk + 65536;  bs = bk + 256;  T = att + 8192;   cs = canon[8 + h]  * inv_sdk; break;
    case 5: Ws = Wv + 65536;  bs = bv + 256;  T = val + 8192;   break;
    case 6: Ws = Wk + 65536;  bs = bk + 256;  T = att + 16384;  cs = canon[16 + h] * inv_sdk; break;
    case 7: Ws = Wv + 65536;  bs = bv + 256;  T = val + 16384;  break;
    case 8: Ws = Wfc;         bs = bfc;       break;
    default: Ws = Wfc + 65536; bs = bfc + 256; break;
  }

  float* Wo = Wtmp + (size_t)p * 65536;
  if (T) {
    float Tcol[32];
#pragma unroll
    for (int d = 0; d < 32; ++d) Tcol[d] = T[h * 1024 + d * 32 + np] * cs;
    for (int k = 0; k < 256; ++k) {
      const float* wr = Ws + (size_t)k * 256 + h * 32;
      float acc = 0.f;
#pragma unroll
      for (int d = 0; d < 32; ++d) acc += wr[d] * Tcol[d];
      Wo[(size_t)k * 256 + n] = acc;
    }
    float bacc = 0.f;
#pragma unroll
    for (int d = 0; d < 32; ++d) bacc += bs[h * 32 + d] * Tcol[d];
    bcomb[p * 256 + n] = bacc;
  } else {
    for (int k = 0; k < 256; ++k) Wo[(size_t)k * 256 + n] = Ws[(size_t)k * 256 + n];
    bcomb[p * 256 + n] = bs[n];
  }
}

// ---------------------------------------------------------------------------
// Kernel 2: pack combined weights into WMMA B-fragment layout (bf16 dwords).
// Layout: [proj][kt(8)][nt(16)][lane(32)][8 dwords].
// lane<16  : N = nt*16+lane,   K = kt*32 + {0..7, 16..23}
// lane>=16 : N = nt*16+lane-16, K = kt*32 + {8..15, 24..31}
// ---------------------------------------------------------------------------
__global__ __launch_bounds__(256)
void pack_frags_kernel(const float* __restrict__ Wtmp, u32* __restrict__ packw) {
  const int gid  = blockIdx.x * 256 + threadIdx.x;   // 10*4096 entries
  const int p    = gid >> 12;
  const int ent  = gid & 4095;
  const int kt   = ent >> 9;
  const int nt   = (ent >> 5) & 15;
  const int lane = ent & 31;
  const int n    = nt * 16 + (lane & 15);
  const int half = lane >> 4;
  const float* W = Wtmp + (size_t)p * 65536;
  u32* out = packw + (size_t)p * 32768 + (size_t)ent * 8;
#pragma unroll
  for (int j = 0; j < 8; ++j) {
    const int k = kt * 32 + ((j < 4) ? 0 : 16) + half * 8 + 2 * (j & 3);
    out[j] = pack_bf2(W[(size_t)k * 256 + n], W[(size_t)(k + 1) * 256 + n]);
  }
}

// ---------------------------------------------------------------------------
// Kernel 3: f32 -> bf16 (packed pairs), optional scale (for 0.5*mean fold).
// ---------------------------------------------------------------------------
__global__ __launch_bounds__(256)
void cast_bf16_kernel(const float* __restrict__ in, u32* __restrict__ out,
                      float scale, long n2) {
  long i = (long)blockIdx.x * 256 + threadIdx.x;
  if (i >= n2) return;
  out[i] = pack_bf2(in[2 * i] * scale, in[2 * i + 1] * scale);
}

__global__ __launch_bounds__(256)
void fill_u32_kernel(u32* __restrict__ p, u32 v, long n) {
  long i = (long)blockIdx.x * 256 + threadIdx.x;
  if (i < n) p[i] = v;
}

// ---------------------------------------------------------------------------
// Kernel 4: WMMA GEMM  C[rows,256] = bf16(A[rows,256]) @ W(256,256) + bias,
// optionally fused sigmoid-gated residual: C = C*alpha + resid*(1-alpha).
// Block = 256 thr (8 waves); W staged fragment-packed in 128 KB LDS via
// CDNA5 async global->LDS B128 copies; each wave computes a 16x256 strip
// (128 v_wmma_f32_16x16x32_bf16).
// ---------------------------------------------------------------------------
__global__ __launch_bounds__(256)
void gemm_wmma_kernel(const u32* __restrict__ Abf,   // [rows*128] bf16-pair dwords
                      const u32* __restrict__ Wpk,   // [32768] packed dwords
                      const float* __restrict__ bias,
                      float* __restrict__ C,
                      const float* __restrict__ resid, // nullable
                      const float* __restrict__ resg,  // nullable (gate logit)
                      int rows) {
  __shared__ u32 lB[32768];                         // 128 KB, LDS offset 0
  const int tid = threadIdx.x;
  {
    // Async cache->LDS staging (GLOBAL_LOAD_ASYNC_TO_LDS_B128, ASYNCcnt).
    // Each lane copies 16B per issue: 32 issues * 256 threads = 128 KB.
    // The LDS-offset operand is derived from the real &lB[...] pointer: the
    // ptrtoint captures lB, so the asm "memory" clobbers are assumed to
    // alias it and the later ds_load fragment reads cannot be undef-folded.
    const char* gbase = (const char*)Wpk;
#pragma unroll
    for (int j = 0; j < 32; ++j) {
      const int i = tid + 256 * j;                  // uint4 index
      u32* lptr = &lB[(size_t)i * 4];               // dword pointer into LDS
      const u32 lds_off = (u32)(uintptr_t)lptr;     // low 32b = LDS byte offset
      const u64 gaddr = (u64)(gbase + (size_t)i * 16u);
      asm volatile("global_load_async_to_lds_b128 %0, %1, off"
                   :: "v"(lds_off), "v"(gaddr)
                   : "memory");
    }
    // Wait for this wave's async copies before any wave reads LDS.
    asm volatile("s_wait_asynccnt 0x0" ::: "memory");
  }
  __syncthreads();

  const int wave = tid >> 5;
  const int lane = tid & 31;
  const int m0 = blockIdx.x * 128 + wave * 16;
  if (m0 >= rows) return;

  const int half = lane >> 4;
  const int l15  = lane & 15;

  // A fragments: row m0+l15, K halves selected by `half` (ISA 16-bit A layout)
  uint4 afrag[8][2];
  {
    const u32* abase = Abf + (size_t)(m0 + l15) * 128;
#pragma unroll
    for (int kt = 0; kt < 8; ++kt) {
      const u32* ap = abase + kt * 16 + half * 4;
      afrag[kt][0] = *(const uint4*)(ap);
      afrag[kt][1] = *(const uint4*)(ap + 8);
    }
  }

  float alpha = 1.0f, beta = 0.0f;
  if (resg) { const float rv = *resg; alpha = 1.0f / (1.0f + __expf(-rv)); beta = 1.0f - alpha; }

#pragma unroll
  for (int nc = 0; nc < 4; ++nc) {
    f32x8 acc[4];
#pragma unroll
    for (int t = 0; t < 4; ++t) {
      const float b = bias[nc * 64 + t * 16 + l15];
      acc[t] = (f32x8){b, b, b, b, b, b, b, b};
    }
#pragma unroll
    for (int kt = 0; kt < 8; ++kt) {
      FragBF a; a.q[0] = afrag[kt][0]; a.q[1] = afrag[kt][1];
#pragma unroll
      for (int t = 0; t < 4; ++t) {
        const int nt = nc * 4 + t;
        const u32* bp = &lB[((kt * 16 + nt) * 32 + lane) * 8];
        FragBF b; b.q[0] = *(const uint4*)(bp); b.q[1] = *(const uint4*)(bp + 4);
        acc[t] = __builtin_amdgcn_wmma_f32_16x16x32_bf16(
            false, a.v, false, b.v, (short)0, acc[t], false, false);
      }
    }
    // C layout: VGPR r, lane<16 -> (m0+r, n0+lane); lane>=16 -> (m0+8+r, ...)
#pragma unroll
    for (int t = 0; t < 4; ++t) {
      const int n = nc * 64 + t * 16 + l15;
#pragma unroll
      for (int r = 0; r < 8; ++r) {
        const int m = m0 + half * 8 + r;
        const size_t idx = (size_t)m * 256 + n;
        float v = acc[t][r];
        if (resid) v = v * alpha + resid[idx] * beta;
        C[idx] = v;
      }
    }
  }
}

// ---------------------------------------------------------------------------
// Edge kernels: score, segment max, exp+denominator, weighted aggregation.
// ---------------------------------------------------------------------------
__global__ __launch_bounds__(256)
void edge_score_kernel(const float* __restrict__ q, const float* __restrict__ ke,
                       const int* __restrict__ src, const int* __restrict__ dst,
                       float* __restrict__ s, int E) {
  const int gid = blockIdx.x * 256 + threadIdx.x;
  const int e = gid >> 3, h = gid & 7;
  if (e >= E) return;
  const float4* qp = (const float4*)(q + (size_t)dst[e] * 256 + h * 32);
  const float4* kp = (const float4*)(ke + (size_t)src[e] * 256 + h * 32);
  float acc = 0.f;
#pragma unroll
  for (int j = 0; j < 8; ++j) {
    const float4 a = qp[j], b = kp[j];
    acc += a.x * b.x + a.y * b.y + a.z * b.z + a.w * b.w;
  }
  s[(size_t)e * 8 + h] = acc;
}

__global__ __launch_bounds__(256)
void seg_max_kernel(const float* __restrict__ s, const int* __restrict__ dst,
                    u32* __restrict__ menc, int E) {
  const int gid = blockIdx.x * 256 + threadIdx.x;
  const int e = gid >> 3, h = gid & 7;
  if (e >= E) return;
  atomicMax(&menc[(size_t)dst[e] * 8 + h], enc_f32(s[(size_t)e * 8 + h]));
}

__global__ __launch_bounds__(256)
void seg_exp_kernel(float* __restrict__ s /* in: score, out: exp */,
                    const int* __restrict__ dst, const u32* __restrict__ menc,
                    float* __restrict__ den, int E) {
  const int gid = blockIdx.x * 256 + threadIdx.x;
  const int e = gid >> 3, h = gid & 7;
  if (e >= E) return;
  const int d = dst[e];
  const float m = dec_f32(menc[(size_t)d * 8 + h]);
  const float v = __expf(s[(size_t)e * 8 + h] - m);
  s[(size_t)e * 8 + h] = v;
  atomicAdd(&den[(size_t)d * 8 + h], v);
}

__global__ __launch_bounds__(256)
void seg_agg_kernel(const float* __restrict__ ve, const float* __restrict__ ex,
                    const float* __restrict__ den, const int* __restrict__ src,
                    const int* __restrict__ dst, float* __restrict__ agg, int E) {
  const int e = blockIdx.x;           // one block per edge
  const int c = threadIdx.x;          // channel 0..255
  if (e >= E) return;
  const int h = c >> 5;
  const int d = dst[e], sr = src[e];
  const float a = ex[(size_t)e * 8 + h] / den[(size_t)d * 8 + h];
  atomicAdd(&agg[(size_t)d * 256 + c], ve[(size_t)sr * 256 + c] * a);
}

// ---------------------------------------------------------------------------
// LayerNorm: one wave per row (32 lanes x 8 channels), shfl_xor reductions.
// ---------------------------------------------------------------------------
__global__ __launch_bounds__(256)
void ln_kernel(const float* __restrict__ hp, const float* __restrict__ g,
               const float* __restrict__ b, float* __restrict__ out, int rows) {
  const int wave = threadIdx.x >> 5, lane = threadIdx.x & 31;
  const int row = blockIdx.x * 8 + wave;
  if (row >= rows) return;
  const float4* rp = (const float4*)(hp + (size_t)row * 256);
  const float4 x0 = rp[lane * 2], x1 = rp[lane * 2 + 1];
  float s = x0.x + x0.y + x0.z + x0.w + x1.x + x1.y + x1.z + x1.w;
  float q = x0.x * x0.x + x0.y * x0.y + x0.z * x0.z + x0.w * x0.w +
            x1.x * x1.x + x1.y * x1.y + x1.z * x1.z + x1.w * x1.w;
#pragma unroll
  for (int o = 16; o > 0; o >>= 1) {
    s += __shfl_xor(s, o, 32);
    q += __shfl_xor(q, o, 32);
  }
  const float mu  = s * (1.0f / 256.0f);
  const float var = q * (1.0f / 256.0f) - mu * mu;
  const float inv = rsqrtf(var + 1e-5f);
  const int c = lane * 8;
  float* op = out + (size_t)row * 256 + c;
  const float xs[8] = {x0.x, x0.y, x0.z, x0.w, x1.x, x1.y, x1.z, x1.w};
#pragma unroll
  for (int j = 0; j < 8; ++j) op[j] = (xs[j] - mu) * inv * g[c + j] + b[c + j];
}

// ---------------------------------------------------------------------------
// Host-side orchestration.
// ---------------------------------------------------------------------------
extern "C" void kernel_launch(void* const* d_in, const int* in_sizes, int n_in,
                              void* d_out, int out_size, void* d_ws, size_t ws_size,
                              hipStream_t stream) {
  const float* h_author = (const float*)d_in[0];
  const float* h_paper  = (const float*)d_in[1];
  const int* writes_src = (const int*)d_in[2];
  const int* writes_dst = (const int*)d_in[3];
  const int* wb_src     = (const int*)d_in[4];
  const int* wb_dst     = (const int*)d_in[5];
  const int* cites_src  = (const int*)d_in[6];
  const int* cites_dst  = (const int*)d_in[7];
  const float* Wk  = (const float*)d_in[8];
  const float* bk  = (const float*)d_in[9];
  const float* Wq  = (const float*)d_in[10];
  const float* bq  = (const float*)d_in[11];
  const float* Wv  = (const float*)d_in[12];
  const float* bv  = (const float*)d_in[13];
  const float* Wfc = (const float*)d_in[14];
  const float* bfc = (const float*)d_in[15];
  const float* ln_g = (const float*)d_in[16];
  const float* ln_b = (const float*)d_in[17];
  const float* att  = (const float*)d_in[18];
  const float* val  = (const float*)d_in[19];
  const float* canon = (const float*)d_in[20];
  const float* res   = (const float*)d_in[21];

  const int NA = in_sizes[0] / 256;
  const int NP = in_sizes[1] / 256;
  const int E  = in_sizes[2];
  float* out = (float*)d_out;

  // ---- workspace layout (256B aligned) ----
  char* ws = (char*)d_ws;
  size_t off = 0;
  auto alloc = [&](size_t bytes) -> char* {
    char* p = ws + off;
    off = (off + bytes + 255) & ~(size_t)255;
    return p;
  };
  float* Wtmp  = (float*)alloc((size_t)10 * 65536 * 4);
  float* bcomb = (float*)alloc((size_t)10 * 256 * 4);
  u32*   packw = (u32*)  alloc((size_t)10 * 32768 * 4);
  u32*   habf  = (u32*)  alloc((size_t)NA * 128 * 4);  // bf16 feats; reused for t_author
  u32*   hpbf  = (u32*)  alloc((size_t)NP * 128 * 4);  // bf16 feats; reused for t_paper
  float* qa    = (float*)alloc((size_t)NA * 256 * 4);  // reused as hp_author
  float* qp    = (float*)alloc((size_t)NP * 256 * 4);  // reused as hp_paper
  float* ke0   = (float*)alloc((size_t)NA * 256 * 4);
  float* ve0   = (float*)alloc((size_t)NA * 256 * 4);
  float* ke1   = (float*)alloc((size_t)NP * 256 * 4);
  float* ve1   = (float*)alloc((size_t)NP * 256 * 4);
  float* ke2   = (float*)alloc((size_t)NP * 256 * 4);
  float* ve2   = (float*)alloc((size_t)NP * 256 * 4);
  float* ex0   = (float*)alloc((size_t)E * 8 * 4);
  float* ex1   = (float*)alloc((size_t)E * 8 * 4);
  float* ex2   = (float*)alloc((size_t)E * 8 * 4);
  u32*   m0e   = (u32*)  alloc((size_t)NP * 8 * 4);
  u32*   m1e   = (u32*)  alloc((size_t)NA * 8 * 4);
  u32*   m2e   = (u32*)  alloc((size_t)NP * 8 * 4);
  float* den0  = (float*)alloc((size_t)NP * 8 * 4);
  float* den1  = (float*)alloc((size_t)NA * 8 * 4);
  float* den2  = (float*)alloc((size_t)NP * 8 * 4);
  float* aggA  = (float*)alloc((size_t)NA * 256 * 4);
  float* aggP  = (float*)alloc((size_t)NP * 256 * 4);
  (void)ws_size; (void)n_in; (void)out_size;

  const u32 ENC_NEG_MAX = 0x00800000u;  // enc_f32(-FLT_MAX)

  // ---- weight prep ----
  build_weights_kernel<<<10, 256, 0, stream>>>(Wq, bq, Wk, bk, Wv, bv, Wfc, bfc,
                                               att, val, canon, Wtmp, bcomb);
  pack_frags_kernel<<<160, 256, 0, stream>>>(Wtmp, packw);

  // ---- feature cast ----
  {
    const long nA2 = (long)NA * 128, nP2 = (long)NP * 128;
    cast_bf16_kernel<<<(int)((nA2 + 255) / 256), 256, 0, stream>>>(h_author, habf, 1.0f, nA2);
    cast_bf16_kernel<<<(int)((nP2 + 255) / 256), 256, 0, stream>>>(h_paper, hpbf, 1.0f, nP2);
  }

  // ---- init accumulators ----
  (void)hipMemsetAsync(den0, 0, (size_t)NP * 8 * 4, stream);
  (void)hipMemsetAsync(den1, 0, (size_t)NA * 8 * 4, stream);
  (void)hipMemsetAsync(den2, 0, (size_t)NP * 8 * 4, stream);
  (void)hipMemsetAsync(aggA, 0, (size_t)NA * 256 * 4, stream);
  (void)hipMemsetAsync(aggP, 0, (size_t)NP * 256 * 4, stream);
  fill_u32_kernel<<<(int)(((long)NP * 8 + 255) / 256), 256, 0, stream>>>(m0e, ENC_NEG_MAX, (long)NP * 8);
  fill_u32_kernel<<<(int)(((long)NA * 8 + 255) / 256), 256, 0, stream>>>(m1e, ENC_NEG_MAX, (long)NA * 8);
  fill_u32_kernel<<<(int)(((long)NP * 8 + 255) / 256), 256, 0, stream>>>(m2e, ENC_NEG_MAX, (long)NP * 8);

  // ---- projection GEMMs (bf16 WMMA) ----
  const int gA = (NA + 127) / 128, gP = (NP + 127) / 128;
  gemm_wmma_kernel<<<gA, 256, 0, stream>>>(habf, packw + 0 * 32768, bcomb + 0 * 256, qa,  nullptr, nullptr, NA);
  gemm_wmma_kernel<<<gP, 256, 0, stream>>>(hpbf, packw + 1 * 32768, bcomb + 1 * 256, qp,  nullptr, nullptr, NP);
  gemm_wmma_kernel<<<gA, 256, 0, stream>>>(habf, packw + 2 * 32768, bcomb + 2 * 256, ke0, nullptr, nullptr, NA);
  gemm_wmma_kernel<<<gA, 256, 0, stream>>>(habf, packw + 3 * 32768, bcomb + 3 * 256, ve0, nullptr, nullptr, NA);
  gemm_wmma_kernel<<<gP, 256, 0, stream>>>(hpbf, packw + 4 * 32768, bcomb + 4 * 256, ke1, nullptr, nullptr, NP);
  gemm_wmma_kernel<<<gP, 256, 0, stream>>>(hpbf, packw + 5 * 32768, bcomb + 5 * 256, ve1, nullptr, nullptr, NP);
  gemm_wmma_kernel<<<gP, 256, 0, stream>>>(hpbf, packw + 6 * 32768, bcomb + 6 * 256, ke2, nullptr, nullptr, NP);
  gemm_wmma_kernel<<<gP, 256, 0, stream>>>(hpbf, packw + 7 * 32768, bcomb + 7 * 256, ve2, nullptr, nullptr, NP);

  // ---- per-edge-type attention (canon/sqrt_dk already folded into ke) ----
  const int gEH = (E * 8 + 255) / 256;
  // writes: author -> paper
  edge_score_kernel<<<gEH, 256, 0, stream>>>(qp, ke0, writes_src, writes_dst, ex0, E);
  seg_max_kernel   <<<gEH, 256, 0, stream>>>(ex0, writes_dst, m0e, E);
  seg_exp_kernel   <<<gEH, 256, 0, stream>>>(ex0, writes_dst, m0e, den0, E);
  seg_agg_kernel   <<<E,   256, 0, stream>>>(ve0, ex0, den0, writes_src, writes_dst, aggP, E);
  // written-by: paper -> author
  edge_score_kernel<<<gEH, 256, 0, stream>>>(qa, ke1, wb_src, wb_dst, ex1, E);
  seg_max_kernel   <<<gEH, 256, 0, stream>>>(ex1, wb_dst, m1e, E);
  seg_exp_kernel   <<<gEH, 256, 0, stream>>>(ex1, wb_dst, m1e, den1, E);
  seg_agg_kernel   <<<E,   256, 0, stream>>>(ve1, ex1, den1, wb_src, wb_dst, aggA, E);
  // cites: paper -> paper (accumulates into aggP; 0.5 mean applied at cast)
  edge_score_kernel<<<gEH, 256, 0, stream>>>(qp, ke2, cites_src, cites_dst, ex2, E);
  seg_max_kernel   <<<gEH, 256, 0, stream>>>(ex2, cites_dst, m2e, E);
  seg_exp_kernel   <<<gEH, 256, 0, stream>>>(ex2, cites_dst, m2e, den2, E);
  seg_agg_kernel   <<<E,   256, 0, stream>>>(ve2, ex2, den2, cites_src, cites_dst, aggP, E);

  // ---- cast aggregates to bf16 (0.5 mean for papers folded into scale) ----
  {
    const long nA2 = (long)NA * 128, nP2 = (long)NP * 128;
    cast_bf16_kernel<<<(int)((nA2 + 255) / 256), 256, 0, stream>>>(aggA, habf, 1.0f, nA2);
    cast_bf16_kernel<<<(int)((nP2 + 255) / 256), 256, 0, stream>>>(aggP, hpbf, 0.5f, nP2);
  }

  // ---- output FC with fused sigmoid-gated residual, then LayerNorm ----
  gemm_wmma_kernel<<<gA, 256, 0, stream>>>(habf, packw + 8 * 32768, bcomb + 8 * 256, qa,
                                           h_author, res + 0, NA);
  gemm_wmma_kernel<<<gP, 256, 0, stream>>>(hpbf, packw + 9 * 32768, bcomb + 9 * 256, qp,
                                           h_paper, res + 1, NP);
  ln_kernel<<<(NA + 7) / 8, 256, 0, stream>>>(qa, ln_g, ln_b, out, NA);
  ln_kernel<<<(NP + 7) / 8, 256, 0, stream>>>(qp, ln_g + 256, ln_b + 256,
                                              out + (size_t)NA * 256, NP);
}
